// AttnEncoder_85152021611174
// MI455X (gfx1250) — matrix operand order repaired and checked by reading, not compiled, not run on hardware
//
#include <hip/hip_runtime.h>
#include <hip/hip_bf16.h>
#include <cstddef>

// ---------------- problem constants ----------------
#define V_  50000
#define D_  300
#define DP_ 320      // D padded to multiple of 32 for WMMA K
#define H_  512
#define G4_ 2048     // 4*H
#define L1_ 256
#define B_  64
#define S_  1024

#define LDS_STRIDE 528   // 512 + 16 bf16 pad (avoids row->same-bank pathology)

// ---------------- WMMA types / helpers ----------------
typedef __attribute__((ext_vector_type(16))) __bf16 v16bf;
typedef __attribute__((ext_vector_type(8)))  __bf16 v8bf;
typedef __attribute__((ext_vector_type(8)))  float  v8f;

#define WMMA_BF16(a, b, c) \
  __builtin_amdgcn_wmma_f32_16x16x32_bf16(false, (a), false, (b), (short)0, (c), false, false)

union FragU { v16bf v; v8bf h[2]; };

// B fragment: 32x16 (KxN) where B = W^T, W row-major [N, ld] (ld == K).
// lane L holds column N = n0 + (L&15); 8 bf16 of W row at k0 + (L>>4)*8 and 8 at +16.
__device__ inline v16bf load_b_frag(const __bf16* __restrict__ W, int ld, int n0, int k0) {
  int lane = threadIdx.x & 31;
  int hi   = lane >> 4;
  const __bf16* p = W + (size_t)(n0 + (lane & 15)) * ld + k0 + hi * 8;
  FragU u;
  u.h[0] = *(const v8bf*)(p);
  u.h[1] = *(const v8bf*)(p + 16);
  return u.v;
}

// A fragment from an LDS row pointer that already folds in (row, hi*8):
// chunk0 at +k0, chunk1 at +k0+16  (A 16x32 MxK layout).
__device__ inline v16bf load_a_lds(const __bf16* arow, int k0) {
  FragU u;
  u.h[0] = *(const v8bf*)(arow + k0);
  u.h[1] = *(const v8bf*)(arow + k0 + 16);
  return u.v;
}

__device__ inline float sigmoidf_(float x) { return 1.f / (1.f + __expf(-x)); }

// ---------------- utility kernels ----------------
__global__ void zero_f32(float* p, int n) {
  int i = blockIdx.x * blockDim.x + threadIdx.x;
  if (i < n) p[i] = 0.f;
}
__global__ void zero_bf16(__bf16* p, int n) {
  int i = blockIdx.x * blockDim.x + threadIdx.x;
  if (i < n) p[i] = (__bf16)0.f;
}
__global__ void add_bias_vec(const float* a, const float* b, float* o, int n) {
  int i = blockIdx.x * blockDim.x + threadIdx.x;
  if (i < n) o[i] = a[i] + b[i];
}
// row-wise f32 -> bf16 with K padding; in: [N, ldin], out: [N, Kpad]
__global__ void convert_pad(const float* __restrict__ in, int ldin, int Kin,
                            __bf16* __restrict__ out, int Kpad) {
  int n = blockIdx.x;
  const float* src = in + (size_t)n * ldin;
  __bf16* dst = out + (size_t)n * Kpad;
  for (int k = threadIdx.x; k < Kpad; k += blockDim.x)
    dst[k] = (k < Kin) ? (__bf16)src[k] : (__bf16)0.f;
}
// embedding gather -> bf16 rows [S*B, 320] (seq-major: row = s*B + b)
__global__ void gather_embed(const int* __restrict__ xi, const float* __restrict__ emb,
                             __bf16* __restrict__ xe) {
  int row = blockIdx.x;
  int s = row >> 6, b = row & 63;
  int idx = xi[b * S_ + s];
  const float* src = emb + (size_t)idx * D_;
  __bf16* dst = xe + (size_t)row * DP_;
  for (int k = threadIdx.x; k < DP_; k += blockDim.x)
    dst[k] = (k < D_) ? (__bf16)src[k] : (__bf16)0.f;
}

// ---------------- generic WMMA GEMM: C[M,N] = A[M,K] * W[N,K]^T (+bias) ----------------
// Block = 8 waves; all waves share 1-2 M-tiles -> A strip staged once in LDS.
// Wave computes 16M x 64N; B fragments double-buffered in registers for latency hiding.
__global__ void wmma_gemm_bias(const __bf16* __restrict__ A, int lda,
                               const __bf16* __restrict__ W, int ldw,
                               const float* __restrict__ bias,
                               float* __restrict__ C, int ldc,
                               int ngroups, int K) {
  __shared__ __align__(16) __bf16 As[32 * LDS_STRIDE];
  int wpb = blockDim.x >> 5;                  // 8 waves
  int gw0 = blockIdx.x * wpb;
  int gw  = gw0 + (threadIdx.x >> 5);
  int mtile  = gw / ngroups;
  int ng     = gw % ngroups;
  int mtile0 = gw0 / ngroups;
  int mtileN = (gw0 + wpb - 1) / ngroups;
  int rows   = (mtileN - mtile0 + 1) * 16;    // 16 or 32

  // cooperative stage of the A strip
  int chunks = K >> 3;                        // 16B chunks per row
  for (int i = threadIdx.x; i < rows * chunks; i += blockDim.x) {
    int r = i / chunks, cc = i - r * chunks;
    *(v8bf*)&As[r * LDS_STRIDE + cc * 8] =
        *(const v8bf*)(A + (size_t)(mtile0 * 16 + r) * lda + cc * 8);
  }
  __syncthreads();

  int lane = threadIdx.x & 31, hi = lane >> 4, col = lane & 15;
  int n0 = ng * 64;
  int lrow0 = (mtile - mtile0) * 16;
  const __bf16* arow = &As[(lrow0 + col) * LDS_STRIDE + hi * 8];

  v8f acc[4] = {};
  v16bf bcur[4];
#pragma unroll
  for (int t = 0; t < 4; ++t) bcur[t] = load_b_frag(W, ldw, n0 + t * 16, 0);

  int k0 = 0;
  for (; k0 + 32 < K; k0 += 32) {
    __builtin_prefetch(W + (size_t)(n0 + col) * ldw + k0 + 96, 0, 3);
    v16bf bn[4];
#pragma unroll
    for (int t = 0; t < 4; ++t) bn[t] = load_b_frag(W, ldw, n0 + t * 16, k0 + 32);
    v16bf a = load_a_lds(arow, k0);
#pragma unroll
    for (int t = 0; t < 4; ++t) acc[t] = WMMA_BF16(a, bcur[t], acc[t]);
#pragma unroll
    for (int t = 0; t < 4; ++t) bcur[t] = bn[t];
  }
  {
    v16bf a = load_a_lds(arow, k0);
#pragma unroll
    for (int t = 0; t < 4; ++t) acc[t] = WMMA_BF16(a, bcur[t], acc[t]);
  }

  int m0 = mtile * 16;
#pragma unroll
  for (int t = 0; t < 4; ++t) {
    int n = n0 + t * 16 + col;
    float bv = bias ? bias[n] : 0.f;
#pragma unroll
    for (int j = 0; j < 8; ++j) {
      int m = m0 + j + 8 * hi;
      C[(size_t)m * ldc + n] = acc[t][j] + bv;
    }
  }
}

// ---------------- LSTM recurrent step (one launch per time step) ----------------
// z = xg[s] + h_prev @ Whh^T ; gates i,f,g,o in N blocks of 512.
// 128 waves: mtile = gw>>5, ntile = gw&31. Block's 8 waves share mtile -> Hprev strip in LDS.
__global__ void lstm_step(const __bf16* __restrict__ Hprev,   // [64,512] bf16
                          const __bf16* __restrict__ Whh,     // [2048,512] bf16
                          const float*  __restrict__ xg,      // [64,2048] f32 (bias included)
                          float*        __restrict__ c,       // [64,512] f32 state
                          __bf16*       __restrict__ hb,      // [64,512] bf16 out
                          float*        __restrict__ hf) {    // [64,512] f32 out (may be null)
  __shared__ __align__(16) __bf16 Hs[16 * LDS_STRIDE];
  int gw = blockIdx.x * 8 + (threadIdx.x >> 5);   // 0..127
  int mtile = gw >> 5, ntile = gw & 31;
  int m0 = mtile * 16, n0 = ntile * 16;

  // stage Hprev strip: 16 rows x 512 = 64 chunks/row
  for (int i = threadIdx.x; i < 16 * 64; i += blockDim.x) {
    int r = i >> 6, cc = i & 63;
    *(v8bf*)&Hs[r * LDS_STRIDE + cc * 8] =
        *(const v8bf*)(Hprev + (size_t)(m0 + r) * H_ + cc * 8);
  }
  __syncthreads();

  int lane = threadIdx.x & 31, hi = lane >> 4, col = lane & 15;
  const __bf16* arow = &Hs[col * LDS_STRIDE + hi * 8];

  v8f ai = {}, af = {}, ag = {}, ao = {};
  v16bf bi = load_b_frag(Whh, H_, n0,          0);
  v16bf bf2= load_b_frag(Whh, H_, n0 + H_,     0);
  v16bf bg = load_b_frag(Whh, H_, n0 + 2 * H_, 0);
  v16bf bo = load_b_frag(Whh, H_, n0 + 3 * H_, 0);

  int k0 = 0;
  for (; k0 + 32 < H_; k0 += 32) {
    v16bf ni = load_b_frag(Whh, H_, n0,          k0 + 32);
    v16bf nf = load_b_frag(Whh, H_, n0 + H_,     k0 + 32);
    v16bf ng = load_b_frag(Whh, H_, n0 + 2 * H_, k0 + 32);
    v16bf no = load_b_frag(Whh, H_, n0 + 3 * H_, k0 + 32);
    v16bf a = load_a_lds(arow, k0);
    ai = WMMA_BF16(a, bi,  ai);
    af = WMMA_BF16(a, bf2, af);
    ag = WMMA_BF16(a, bg,  ag);
    ao = WMMA_BF16(a, bo,  ao);
    bi = ni; bf2 = nf; bg = ng; bo = no;
  }
  {
    v16bf a = load_a_lds(arow, k0);
    ai = WMMA_BF16(a, bi,  ai);
    af = WMMA_BF16(a, bf2, af);
    ag = WMMA_BF16(a, bg,  ag);
    ao = WMMA_BF16(a, bo,  ao);
  }

  int n = n0 + col;
#pragma unroll
  for (int j = 0; j < 8; ++j) {
    int m = m0 + j + 8 * hi;
    const float* xr = xg + (size_t)m * G4_;
    float zi = ai[j] + xr[n];
    float zf = af[j] + xr[H_ + n];
    float zg = ag[j] + xr[2 * H_ + n];
    float zo = ao[j] + xr[3 * H_ + n];
    size_t idx = (size_t)m * H_ + n;
    float cn = sigmoidf_(zf) * c[idx] + sigmoidf_(zi) * tanhf(zg);
    float hn = sigmoidf_(zo) * tanhf(cn);
    c[idx]  = cn;
    hb[idx] = (__bf16)hn;
    if (hf) hf[idx] = hn;
  }
}

// ---------------- attention: t = tanh(e1[b] + h_t @ W1b^T + b1) ----------------
// Same structure as generic GEMM (ngroups=4 -> 2 M-tiles per block staged in LDS).
__global__ void attn_t_kernel(const __bf16* __restrict__ A,    // h2 bf16 [S*B, 512]
                              const __bf16* __restrict__ W,    // W1b bf16 [256,512]
                              const float*  __restrict__ e1,   // [64,256]
                              const float*  __restrict__ b1,   // [256]
                              float*        __restrict__ T) {  // [S*B, 256]
  __shared__ __align__(16) __bf16 As[32 * LDS_STRIDE];
  int gw0 = blockIdx.x * 8;
  int gw  = gw0 + (threadIdx.x >> 5);
  int mtile = gw >> 2, ng = gw & 3;
  int mtile0 = gw0 >> 2;                  // 2 m-tiles per block
  for (int i = threadIdx.x; i < 32 * 64; i += blockDim.x) {
    int r = i >> 6, cc = i & 63;
    *(v8bf*)&As[r * LDS_STRIDE + cc * 8] =
        *(const v8bf*)(A + (size_t)(mtile0 * 16 + r) * H_ + cc * 8);
  }
  __syncthreads();

  int lane = threadIdx.x & 31, hi = lane >> 4, col = lane & 15;
  int n0 = ng * 64;
  const __bf16* arow = &As[((mtile - mtile0) * 16 + col) * LDS_STRIDE + hi * 8];

  v8f acc[4] = {};
  v16bf bcur[4];
#pragma unroll
  for (int t = 0; t < 4; ++t) bcur[t] = load_b_frag(W, H_, n0 + t * 16, 0);
  int k0 = 0;
  for (; k0 + 32 < H_; k0 += 32) {
    v16bf bn[4];
#pragma unroll
    for (int t = 0; t < 4; ++t) bn[t] = load_b_frag(W, H_, n0 + t * 16, k0 + 32);
    v16bf a = load_a_lds(arow, k0);
#pragma unroll
    for (int t = 0; t < 4; ++t) acc[t] = WMMA_BF16(a, bcur[t], acc[t]);
#pragma unroll
    for (int t = 0; t < 4; ++t) bcur[t] = bn[t];
  }
  {
    v16bf a = load_a_lds(arow, k0);
#pragma unroll
    for (int t = 0; t < 4; ++t) acc[t] = WMMA_BF16(a, bcur[t], acc[t]);
  }

  int m0 = mtile * 16;
#pragma unroll
  for (int t = 0; t < 4; ++t) {
    int n = n0 + t * 16 + col;
#pragma unroll
    for (int j = 0; j < 8; ++j) {
      int m = m0 + j + 8 * hi;
      int b = m & 63;
      T[(size_t)m * L1_ + n] = tanhf(acc[t][j] + e1[b * L1_ + n] + b1[n]);
    }
  }
}

// e[b,s] = dot(t[s*B+b,:], W2) + b2 ; one wave per row
__global__ void e_reduce(const float* __restrict__ T, const float* __restrict__ W2,
                         const float* __restrict__ b2, float* __restrict__ e) {
  int row  = blockIdx.x * (blockDim.x >> 5) + (threadIdx.x >> 5);
  int lane = threadIdx.x & 31;
  const float* tr = T + (size_t)row * L1_;
  float acc = 0.f;
  for (int k = lane; k < L1_; k += 32) acc += tr[k] * W2[k];
  for (int off = 16; off; off >>= 1) acc += __shfl_down(acc, off, 32);
  if (lane == 0) {
    int s = row >> 6, b = row & 63;
    e[b * S_ + s] = acc + b2[0];
  }
}

__global__ void softmax_rows(const float* __restrict__ e, float* __restrict__ attn) {
  __shared__ float red[256];
  int b = blockIdx.x;
  const float* er = e + (size_t)b * S_;
  float m = -1e30f;
  for (int s = threadIdx.x; s < S_; s += 256) m = fmaxf(m, er[s]);
  red[threadIdx.x] = m; __syncthreads();
  for (int off = 128; off; off >>= 1) {
    if (threadIdx.x < off) red[threadIdx.x] = fmaxf(red[threadIdx.x], red[threadIdx.x + off]);
    __syncthreads();
  }
  m = red[0]; __syncthreads();
  float sum = 0.f;
  for (int s = threadIdx.x; s < S_; s += 256) sum += __expf(er[s] - m);
  red[threadIdx.x] = sum; __syncthreads();
  for (int off = 128; off; off >>= 1) {
    if (threadIdx.x < off) red[threadIdx.x] += red[threadIdx.x + off];
    __syncthreads();
  }
  float inv = 1.f / red[0];
  for (int s = threadIdx.x; s < S_; s += 256)
    attn[(size_t)b * S_ + s] = __expf(er[s] - m) * inv;
}

// context[b,h] = sum_s attn[b,s] * h2[s,b,h]
__global__ void context_kernel(const float* __restrict__ attn, const float* __restrict__ h2f,
                               float* __restrict__ ctx) {
  int b = blockIdx.y;
  int h = blockIdx.x * 256 + threadIdx.x;
  const float* ar = attn + (size_t)b * S_;
  float acc = 0.f;
  for (int s = 0; s < S_; ++s) acc += ar[s] * h2f[((size_t)s * B_ + b) * H_ + h];
  ctx[(size_t)b * H_ + h] = acc;
}

// out[b] = sigmoid([context, h_N] . Wf + bf)
__global__ void final_kernel(const float* __restrict__ ctx, const float* __restrict__ hN,
                             const float* __restrict__ Wf, const float* __restrict__ bfp,
                             float* __restrict__ out) {
  __shared__ float red[256];
  int b = blockIdx.x;
  float acc = 0.f;
  for (int h = threadIdx.x; h < H_; h += 256)
    acc += ctx[(size_t)b * H_ + h] * Wf[h] + hN[(size_t)b * H_ + h] * Wf[H_ + h];
  red[threadIdx.x] = acc; __syncthreads();
  for (int off = 128; off; off >>= 1) {
    if (threadIdx.x < off) red[threadIdx.x] += red[threadIdx.x + off];
    __syncthreads();
  }
  if (threadIdx.x == 0) out[b] = 1.f / (1.f + __expf(-(red[0] + bfp[0])));
}

// ---------------- host orchestration ----------------
extern "C" void kernel_launch(void* const* d_in, const int* in_sizes, int n_in,
                              void* d_out, int out_size, void* d_ws, size_t ws_size,
                              hipStream_t stream) {
  const int*   x_index = (const int*)  d_in[0];
  const float* embed   = (const float*)d_in[1];
  const float* Wih0    = (const float*)d_in[2];
  const float* Whh0    = (const float*)d_in[3];
  const float* bih0    = (const float*)d_in[4];
  const float* bhh0    = (const float*)d_in[5];
  const float* Wih1    = (const float*)d_in[6];
  const float* Whh1    = (const float*)d_in[7];
  const float* bih1    = (const float*)d_in[8];
  const float* bhh1    = (const float*)d_in[9];
  const float* W1      = (const float*)d_in[10];
  const float* b1      = (const float*)d_in[11];
  const float* W2      = (const float*)d_in[12];
  const float* b2      = (const float*)d_in[13];
  const float* Wf      = (const float*)d_in[14];
  const float* bfin    = (const float*)d_in[15];

  float* out_attn = (float*)d_out;                    // [B,S]
  float* out_o    = (float*)d_out + (size_t)B_ * S_;  // [B]

  // carve workspace
  char* base = (char*)d_ws;
  size_t off = 0;
  auto take = [&](size_t bytes) -> char* {
    char* p = base + off;
    off += (bytes + 255) & ~(size_t)255;
    return p;
  };
  const size_t SB = (size_t)S_ * B_;
  float*  xg     = (float*) take(SB * G4_ * 4);   // [S*B, 2048] gate preacts (reused L0/L1)
  float*  h2f    = (float*) take(SB * H_ * 4);    // layer-1 h, f32
  __bf16* h1b    = (__bf16*)take(SB * H_ * 2);    // layer-0 h, bf16
  __bf16* h2b    = (__bf16*)take(SB * H_ * 2);    // layer-1 h, bf16
  __bf16* xe     = (__bf16*)take(SB * DP_ * 2);   // gathered embeddings, bf16, K-padded
  float*  tbuf   = (float*) take(SB * L1_ * 4);   // tanh(...) attention hidden
  float*  ebuf   = (float*) take((size_t)B_ * S_ * 4);
  __bf16* Wb_ih0 = (__bf16*)take((size_t)G4_ * DP_ * 2);
  __bf16* Wb_hh0 = (__bf16*)take((size_t)G4_ * H_ * 2);
  __bf16* Wb_ih1 = (__bf16*)take((size_t)G4_ * H_ * 2);
  __bf16* Wb_hh1 = (__bf16*)take((size_t)G4_ * H_ * 2);
  __bf16* W1a_b  = (__bf16*)take((size_t)L1_ * H_ * 2);
  __bf16* W1b_b  = (__bf16*)take((size_t)L1_ * H_ * 2);
  float*  bias0  = (float*) take(G4_ * 4);
  float*  bias1  = (float*) take(G4_ * 4);
  float*  cstate = (float*) take((size_t)B_ * H_ * 4);
  __bf16* hzero  = (__bf16*)take((size_t)B_ * H_ * 2);
  float*  e1buf  = (float*) take((size_t)B_ * L1_ * 4);
  float*  ctx    = (float*) take((size_t)B_ * H_ * 4);

  // --- prep: weight conversion, bias merge, zero init ---
  convert_pad<<<G4_, 256, 0, stream>>>(Wih0, D_, D_, Wb_ih0, DP_);
  convert_pad<<<G4_, 256, 0, stream>>>(Whh0, H_, H_, Wb_hh0, H_);
  convert_pad<<<G4_, 256, 0, stream>>>(Wih1, H_, H_, Wb_ih1, H_);
  convert_pad<<<G4_, 256, 0, stream>>>(Whh1, H_, H_, Wb_hh1, H_);
  convert_pad<<<L1_, 256, 0, stream>>>(W1,      2 * H_, H_, W1a_b, H_);
  convert_pad<<<L1_, 256, 0, stream>>>(W1 + H_, 2 * H_, H_, W1b_b, H_);
  add_bias_vec<<<8, 256, 0, stream>>>(bih0, bhh0, bias0, G4_);
  add_bias_vec<<<8, 256, 0, stream>>>(bih1, bhh1, bias1, G4_);
  zero_bf16<<<(B_ * H_) / 256, 256, 0, stream>>>(hzero, B_ * H_);

  // --- embedding gather (seq-major) ---
  gather_embed<<<(int)SB, 64, 0, stream>>>(x_index, embed, xe);

  // --- layer 0 input projection: xg = xe @ Wih0^T + bias0 ---
  wmma_gemm_bias<<<16384, 256, 0, stream>>>(xe, DP_, Wb_ih0, DP_, bias0, xg, G4_, G4_ / 64, DP_);

  // --- layer 0 recurrence ---
  zero_f32<<<(B_ * H_) / 256, 256, 0, stream>>>(cstate, B_ * H_);
  for (int s = 0; s < S_; ++s) {
    const __bf16* hp = s ? (h1b + (size_t)(s - 1) * B_ * H_) : hzero;
    lstm_step<<<16, 256, 0, stream>>>(hp, Wb_hh0, xg + (size_t)s * B_ * G4_, cstate,
                                      h1b + (size_t)s * B_ * H_, nullptr);
  }

  // --- layer 1 input projection: xg = h1 @ Wih1^T + bias1 ---
  wmma_gemm_bias<<<16384, 256, 0, stream>>>(h1b, H_, Wb_ih1, H_, bias1, xg, G4_, G4_ / 64, H_);

  // --- layer 1 recurrence ---
  zero_f32<<<(B_ * H_) / 256, 256, 0, stream>>>(cstate, B_ * H_);
  for (int s = 0; s < S_; ++s) {
    const __bf16* hp = s ? (h2b + (size_t)(s - 1) * B_ * H_) : hzero;
    lstm_step<<<16, 256, 0, stream>>>(hp, Wb_hh1, xg + (size_t)s * B_ * G4_, cstate,
                                      h2b + (size_t)s * B_ * H_, h2f + (size_t)s * B_ * H_);
  }

  // --- attention ---
  const __bf16* hNb = h2b + (size_t)(S_ - 1) * B_ * H_;
  const float*  hNf = h2f + (size_t)(S_ - 1) * B_ * H_;
  // e1 = h_N @ W1a^T : M=64, N=256, K=512 -> 16 waves = 2 blocks
  wmma_gemm_bias<<<2, 256, 0, stream>>>(hNb, H_, W1a_b, H_, nullptr, e1buf, L1_, L1_ / 64, H_);
  // t = tanh(e1[b] + h_t @ W1b^T + b1) : M=65536, N=256, K=512 -> 2048 blocks
  attn_t_kernel<<<2048, 256, 0, stream>>>(h2b, W1b_b, e1buf, b1, tbuf);
  // e = t . W2 + b2
  e_reduce<<<(int)(SB / 8), 256, 0, stream>>>(tbuf, W2, b2, ebuf);
  // softmax over S -> attn_weights (output 0)
  softmax_rows<<<B_, 256, 0, stream>>>(ebuf, out_attn);
  // context
  context_kernel<<<dim3(H_ / 256, B_), 256, 0, stream>>>(out_attn, h2f, ctx);
  // head (output 1)
  final_kernel<<<B_, 256, 0, stream>>>(ctx, hNf, Wf, bfin, out_o);
}
